// Perception_12807592477478
// MI455X (gfx1250) — matrix operand order repaired and checked by reading, not compiled, not run on hardware
//
#include <hip/hip_runtime.h>
#include <hip/hip_bf16.h>

#define BB 16
#define NN 512
#define DD 512
#define HH 8
#define FFD 2048

typedef __attribute__((ext_vector_type(16))) __bf16 v16bf;
typedef __attribute__((ext_vector_type(8)))  float  v8f;
typedef __attribute__((ext_vector_type(8)))  unsigned int v8u;

__device__ __forceinline__ unsigned short f2bf(float f) {
    unsigned int u = __float_as_uint(f);
    u += 0x7FFFu + ((u >> 16) & 1u);          // round-to-nearest-even
    return (unsigned short)(u >> 16);
}
__device__ __forceinline__ float bf2f(unsigned short h) {
    return __uint_as_float(((unsigned int)h) << 16);
}
// Sinusoidal PE value, pe[t, d]
__device__ __forceinline__ float pe_val(int rel, int d) {
    const float c = -9.210340371976184f / 512.0f;  // -ln(10000)/D
    float dv = __expf(c * (float)(d & ~1));
    float a = (float)rel * dv;
    return (d & 1) ? __cosf(a) : __sinf(a);
}
__device__ __forceinline__ v16bf mk_frag(uint4 lo, uint4 hi) {
    v8u u = { lo.x, lo.y, lo.z, lo.w, hi.x, hi.y, hi.z, hi.w };
    return __builtin_bit_cast(v16bf, u);
}

// ---------------------------------------------------------------------------
// bf16 WMMA GEMM:  C[m,n] = act( sum_k A[m,k]*Bt[n,k] + bias[n] )
// A: bf16 row-major [M,K] (lda), Bt: bf16 col-major-of-B i.e. [N,K] (ldb)
// Optional bf16 output (normal or transposed) and/or f32 output.
// Block = 256 threads = 8 waves tiled 4(M) x 2(N); wave does 16x64.
// K-loop is software-pipelined: next step's fragments prefetched while the
// current step's 4 WMMAs execute (overlaps global-load latency with math).
// Grid: (M/64, N/128, batches)
// ---------------------------------------------------------------------------
__global__ __launch_bounds__(256)
void gemm_wmma_kernel(const unsigned short* __restrict__ A, int lda, long sA,
                      const unsigned short* __restrict__ Bt, int ldb, long sB,
                      const float* __restrict__ bias,
                      unsigned short* __restrict__ Cb, int ldcb, long sCb, int transC,
                      float* __restrict__ Cf, int ldcf, long sCf,
                      int K, int relu)
{
    int z = blockIdx.z;
    A  += (long)z * sA;
    Bt += (long)z * sB;
    if (Cb) Cb += (long)z * sCb;
    if (Cf) Cf += (long)z * sCf;

    int wave = threadIdx.x >> 5;     // 0..7
    int lane = threadIdx.x & 31;
    int wm   = wave >> 1;            // 0..3
    int wn   = wave & 1;             // 0..1
    int m0   = blockIdx.x * 64  + wm * 16;
    int n0   = blockIdx.y * 128 + wn * 64;
    int half = lane >> 4;            // 0/1
    int l16  = lane & 15;

    v8f acc[4];
#pragma unroll
    for (int t = 0; t < 4; ++t)
#pragma unroll
        for (int r = 0; r < 8; ++r) acc[t][r] = 0.0f;

    // A fragment rows: 16x32, contiguous k order matching B (lanes 0-15: K=0..15; 16-31: K=16..31)
    const unsigned short* arow = A + (long)(m0 + l16) * lda + half * 16;
    // B fragment cols: 32x16, lane = column; lanes 0-15 K=0..15, 16-31 K=16..31
    const unsigned short* bptr[4];
#pragma unroll
    for (int t = 0; t < 4; ++t)
        bptr[t] = Bt + (long)(n0 + t * 16 + l16) * ldb + half * 16;

    // ---- prologue: load k0 = 0 fragments ----
    uint4 aLo = *(const uint4*)(arow);
    uint4 aHi = *(const uint4*)(arow + 8);
    uint4 bLo[4], bHi[4];
#pragma unroll
    for (int t = 0; t < 4; ++t) {
        bLo[t] = *(const uint4*)(bptr[t]);
        bHi[t] = *(const uint4*)(bptr[t] + 8);
    }

    // ---- pipelined main loop ----
    for (int k0 = 0; k0 < K - 32; k0 += 32) {
        int kn = k0 + 32;
        uint4 naLo = *(const uint4*)(arow + kn);
        uint4 naHi = *(const uint4*)(arow + kn + 8);
        uint4 nbLo[4], nbHi[4];
#pragma unroll
        for (int t = 0; t < 4; ++t) {
            nbLo[t] = *(const uint4*)(bptr[t] + kn);
            nbHi[t] = *(const uint4*)(bptr[t] + kn + 8);
        }
        v16bf af = mk_frag(aLo, aHi);
#pragma unroll
        for (int t = 0; t < 4; ++t) {
            acc[t] = __builtin_amdgcn_wmma_f32_16x16x32_bf16(
                false, af, false, mk_frag(bLo[t], bHi[t]), (short)0, acc[t], false, false);
        }
        aLo = naLo; aHi = naHi;
#pragma unroll
        for (int t = 0; t < 4; ++t) { bLo[t] = nbLo[t]; bHi[t] = nbHi[t]; }
    }
    // ---- epilogue K-step ----
    {
        v16bf af = mk_frag(aLo, aHi);
#pragma unroll
        for (int t = 0; t < 4; ++t) {
            acc[t] = __builtin_amdgcn_wmma_f32_16x16x32_bf16(
                false, af, false, mk_frag(bLo[t], bHi[t]), (short)0, acc[t], false, false);
        }
    }

#pragma unroll
    for (int t = 0; t < 4; ++t) {
        int n = n0 + t * 16 + l16;
        float bv = bias ? bias[n] : 0.0f;
#pragma unroll
        for (int r = 0; r < 8; ++r) {
            int m = m0 + r + half * 8;          // C layout: VGPR r -> M=r(+8)
            float v = acc[t][r] + bv;
            if (relu) v = fmaxf(v, 0.0f);
            if (Cf) Cf[(long)m * ldcf + n] = v;
            if (Cb) {
                if (transC) Cb[(long)n * ldcb + m] = f2bf(v);
                else        Cb[(long)m * ldcb + n] = f2bf(v);
            }
        }
    }
}

// ---------------------------------------------------------------------------
// Elementwise / prep kernels
// ---------------------------------------------------------------------------
__global__ __launch_bounds__(256)
void cvt_wt_kernel(const float* __restrict__ W, unsigned short* __restrict__ Wt,
                   int K, int N)   // Wt[n*K+k] = bf16(W[k*N+n])
{
    long idx = (long)blockIdx.x * 256 + threadIdx.x;
    if (idx >= (long)K * N) return;
    int k = (int)(idx / N), n = (int)(idx % N);
    Wt[(long)n * K + k] = f2bf(W[idx]);
}

__global__ __launch_bounds__(256)
void build_hcat_kernel(const float* __restrict__ gm, const int* __restrict__ gtime,
                       const int* __restrict__ step, const float* __restrict__ goal,
                       unsigned short* __restrict__ hcat)
{
    long idx = (long)blockIdx.x * 256 + threadIdx.x;   // over B*N*D
    if (idx >= (long)BB * NN * DD) return;
    int d = (int)(idx & (DD - 1));
    long bn = idx >> 9;
    int b = (int)(bn >> 9);
    int rel = step[b] - gtime[bn];
    hcat[bn * 1024 + d]       = f2bf(gm[idx] + pe_val(rel, d));
    hcat[bn * 1024 + 512 + d] = f2bf(goal[b * DD + d]);
}

__global__ __launch_bounds__(256)
void degree_kernel(const float* __restrict__ A, float* __restrict__ r)
{
    int row = blockIdx.x;                  // b*N + j ; A symmetric -> row sum
    const float* arow = A + (long)row * NN;
    float s = 0.0f;
    for (int i = threadIdx.x; i < NN; i += 256) s += arow[i];
    __shared__ float red[256];
    red[threadIdx.x] = s; __syncthreads();
    for (int st = 128; st > 0; st >>= 1) {
        if (threadIdx.x < st) red[threadIdx.x] += red[threadIdx.x + st];
        __syncthreads();
    }
    if (threadIdx.x == 0) r[row] = rsqrtf(red[0] + 1.0f);   // +I diagonal
}

__global__ __launch_bounds__(256)
void build_norma_kernel(const float* __restrict__ A, const float* __restrict__ r,
                        unsigned short* __restrict__ nA)
{
    long idx = (long)blockIdx.x * 256 + threadIdx.x;   // over B*N*N
    if (idx >= (long)BB * NN * NN) return;
    int b = (int)(idx >> 18);
    int rem = (int)(idx & (NN * NN - 1));
    int i = rem >> 9, j = rem & (NN - 1);
    float a = A[idx] + (i == j ? 1.0f : 0.0f);
    nA[idx] = f2bf(r[b * NN + i] * r[b * NN + j] * a);
}

__global__ __launch_bounds__(256)
void gcpe_kernel(const float* __restrict__ gc, const int* __restrict__ gtime,
                 const int* __restrict__ step, unsigned short* __restrict__ gcpe)
{
    long idx = (long)blockIdx.x * 256 + threadIdx.x;   // over B*N*D
    if (idx >= (long)BB * NN * DD) return;
    int d = (int)(idx & (DD - 1));
    long bn = idx >> 9;
    int b = (int)(bn >> 9);
    int rel = step[b] - gtime[bn];
    gcpe[idx] = f2bf(gc[idx] + pe_val(rel, d));
}

// ---------------------------------------------------------------------------
// Folded attention (q_len == 1)
// ---------------------------------------------------------------------------
__global__ __launch_bounds__(256)
void qproj_kernel(const float* __restrict__ src, const float* __restrict__ wq,
                  const float* __restrict__ bq, float* __restrict__ Qh)
{
    int idx = blockIdx.x * 256 + threadIdx.x;   // B*D
    int b = idx >> 9, o = idx & (DD - 1);
    float acc = bq[o];
    const float* s = src + b * DD;
    for (int d = 0; d < DD; ++d) acc += s[d] * wq[(long)d * DD + o];
    Qh[idx] = acc;
}

__global__ __launch_bounds__(256)
void uproj_kernel(const float* __restrict__ wk, const float* __restrict__ Qh,
                  float* __restrict__ u)       // u[b,h,d] = sum_e wk[d, h*64+e]*Qh[b,h*64+e]
{
    int idx = blockIdx.x * 256 + threadIdx.x;  // B*H*D
    int bh = idx >> 9, d = idx & (DD - 1);
    int b = bh >> 3, h = bh & 7;
    const float* q = Qh + b * DD + h * 64;
    const float* w = wk + (long)d * DD + h * 64;
    float acc = 0.0f;
    for (int e = 0; e < 64; ++e) acc += w[e] * q[e];
    u[idx] = acc;
}

__global__ void cb_kernel(const float* __restrict__ Qh, const float* __restrict__ bk,
                          float* __restrict__ cb)
{
    int idx = threadIdx.x;                     // B*H = 128
    int b = idx >> 3, h = idx & 7;
    float acc = 0.0f;
    for (int e = 0; e < 64; ++e) acc += Qh[b * DD + h * 64 + e] * bk[h * 64 + e];
    cb[idx] = acc;
}

__global__ __launch_bounds__(256)
void scores_softmax_kernel(const unsigned short* __restrict__ gcpe,
                           const float* __restrict__ u, const float* __restrict__ cb,
                           const int* __restrict__ mask, float* __restrict__ attnw)
{
    int bh = blockIdx.x;                       // B*H blocks
    int b = bh >> 3;
    __shared__ float uS[DD];
    __shared__ float red[256];
    for (int d = threadIdx.x; d < DD; d += 256) uS[d] = u[(long)bh * DD + d];
    __syncthreads();
    float sc[2];
#pragma unroll
    for (int t = 0; t < 2; ++t) {
        int j = threadIdx.x + t * 256;
        const unsigned short* g = gcpe + ((long)b * NN + j) * DD;
        float acc = 0.0f;
        for (int d = 0; d < DD; ++d) acc += bf2f(g[d]) * uS[d];
        acc = (acc + cb[bh]) * 0.125f;         // / sqrt(64)
        if (mask[b * NN + j] == 0) acc = -1e30f;
        sc[t] = acc;
    }
    red[threadIdx.x] = fmaxf(sc[0], sc[1]); __syncthreads();
    for (int st = 128; st > 0; st >>= 1) {
        if (threadIdx.x < st) red[threadIdx.x] = fmaxf(red[threadIdx.x], red[threadIdx.x + st]);
        __syncthreads();
    }
    float mx = red[0]; __syncthreads();
    float e0 = __expf(sc[0] - mx), e1 = __expf(sc[1] - mx);
    red[threadIdx.x] = e0 + e1; __syncthreads();
    for (int st = 128; st > 0; st >>= 1) {
        if (threadIdx.x < st) red[threadIdx.x] += red[threadIdx.x + st];
        __syncthreads();
    }
    float inv = 1.0f / red[0];
    attnw[(long)bh * NN + threadIdx.x]       = e0 * inv;
    attnw[(long)bh * NN + threadIdx.x + 256] = e1 * inv;
}

__global__ __launch_bounds__(256)
void ctx_kernel(const float* __restrict__ attnw, const unsigned short* __restrict__ gcpe,
                float* __restrict__ wctx)      // wctx[b,h,d] = sum_j attn[b,h,j]*gcpe[b,j,d]
{
    int idx = blockIdx.x * 256 + threadIdx.x;  // B*H*D
    int bh = idx >> 9, d = idx & (DD - 1);
    int b = bh >> 3;
    const float* aw = attnw + (long)bh * NN;
    const unsigned short* g = gcpe + (long)b * NN * DD + d;
    float acc = 0.0f;
    for (int j = 0; j < NN; ++j) acc += aw[j] * bf2f(g[(long)j * DD]);
    wctx[idx] = acc;
}

__global__ __launch_bounds__(256)
void o1_kernel(const float* __restrict__ wctx, const float* __restrict__ wv,
               const float* __restrict__ bv, float* __restrict__ o1)
{
    int idx = blockIdx.x * 256 + threadIdx.x;  // B*D
    int b = idx >> 9, n = idx & (DD - 1);
    int h = n >> 6;
    const float* w = wctx + ((long)b * HH + h) * DD;
    float acc = bv[n];
    for (int d = 0; d < DD; ++d) acc += w[d] * wv[(long)d * DD + n];
    o1[idx] = acc;
}

__global__ __launch_bounds__(256)
void o2_resid_kernel(const float* __restrict__ o1, const float* __restrict__ wo,
                     const float* __restrict__ bo, const float* __restrict__ src,
                     float* __restrict__ o2)
{
    int idx = blockIdx.x * 256 + threadIdx.x;  // B*D
    int b = idx >> 9, i = idx & (DD - 1);
    float acc = bo[i] + src[idx];
    const float* o = o1 + b * DD;
    for (int n = 0; n < DD; ++n) acc += o[n] * wo[(long)n * DD + i];
    o2[idx] = acc;
}

__global__ __launch_bounds__(256)
void ln_kernel(const float* __restrict__ x, const float* __restrict__ s,
               const float* __restrict__ bb, float* __restrict__ out)
{
    int b = blockIdx.x;
    const float* xv = x + b * DD;
    __shared__ float red[256];
    float v0 = xv[threadIdx.x], v1 = xv[threadIdx.x + 256];
    red[threadIdx.x] = v0 + v1; __syncthreads();
    for (int st = 128; st > 0; st >>= 1) {
        if (threadIdx.x < st) red[threadIdx.x] += red[threadIdx.x + st];
        __syncthreads();
    }
    float mean = red[0] * (1.0f / DD); __syncthreads();
    float d0 = v0 - mean, d1 = v1 - mean;
    red[threadIdx.x] = d0 * d0 + d1 * d1; __syncthreads();
    for (int st = 128; st > 0; st >>= 1) {
        if (threadIdx.x < st) red[threadIdx.x] += red[threadIdx.x + st];
        __syncthreads();
    }
    float inv = rsqrtf(red[0] * (1.0f / DD) + 1e-5f);
    out[b * DD + threadIdx.x]       = d0 * inv * s[threadIdx.x]       + bb[threadIdx.x];
    out[b * DD + threadIdx.x + 256] = d1 * inv * s[threadIdx.x + 256] + bb[threadIdx.x + 256];
}

__global__ __launch_bounds__(256)
void ff1_kernel(const float* __restrict__ xln, const float* __restrict__ l1w,
                const float* __restrict__ l1b, float* __restrict__ ff1)
{
    int idx = blockIdx.x * 256 + threadIdx.x;  // B*FF
    int b = idx >> 11, j = idx & (FFD - 1);
    float acc = l1b[j];
    const float* x = xln + b * DD;
    for (int i = 0; i < DD; ++i) acc += x[i] * l1w[(long)i * FFD + j];
    ff1[idx] = fmaxf(acc, 0.0f);
}

__global__ __launch_bounds__(256)
void ff2_resid_kernel(const float* __restrict__ ff1, const float* __restrict__ l2w,
                      const float* __restrict__ l2b, const float* __restrict__ xln,
                      float* __restrict__ y)
{
    int idx = blockIdx.x * 256 + threadIdx.x;  // B*D
    int b = idx >> 9, i = idx & (DD - 1);
    float acc = l2b[i] + xln[idx];
    const float* f = ff1 + (long)b * FFD;
    for (int j = 0; j < FFD; ++j) acc += f[j] * l2w[(long)j * DD + i];
    y[idx] = acc;
}

// ---------------------------------------------------------------------------
// Host side
// ---------------------------------------------------------------------------
static void launch_gemm(const unsigned short* A, int lda, long sA,
                        const unsigned short* Bt, int ldb, long sB,
                        const float* bias,
                        unsigned short* Cb, int ldcb, long sCb, int transC,
                        float* Cf, int ldcf, long sCf,
                        int M, int N, int K, int relu, int batches, hipStream_t st)
{
    dim3 grid(M / 64, N / 128, batches), block(256);
    gemm_wmma_kernel<<<grid, block, 0, st>>>(A, lda, sA, Bt, ldb, sB, bias,
                                             Cb, ldcb, sCb, transC, Cf, ldcf, sCf, K, relu);
}

static void run_decoder(const float* src, void* const* din, int base,
                        const unsigned short* gcpe, const int* mask,
                        float* out, char* scr, hipStream_t st)
{
    const float* wq   = (const float*)din[base + 0];
    const float* bq   = (const float*)din[base + 1];
    const float* wk   = (const float*)din[base + 2];
    const float* bk   = (const float*)din[base + 3];
    const float* wv   = (const float*)din[base + 4];
    const float* bv   = (const float*)din[base + 5];
    const float* wo   = (const float*)din[base + 6];
    const float* bo   = (const float*)din[base + 7];
    const float* l1w  = (const float*)din[base + 8];
    const float* l1b  = (const float*)din[base + 9];
    const float* l2w  = (const float*)din[base + 10];
    const float* l2b  = (const float*)din[base + 11];
    const float* ln1s = (const float*)din[base + 12];
    const float* ln1b = (const float*)din[base + 13];
    const float* ln2s = (const float*)din[base + 14];
    const float* ln2b = (const float*)din[base + 15];

    size_t o = 0;
    float* Qh    = (float*)(scr + o); o += 32768;
    float* u     = (float*)(scr + o); o += 262144;
    float* cb    = (float*)(scr + o); o += 4096;
    float* attnw = (float*)(scr + o); o += 262144;
    float* wctx  = (float*)(scr + o); o += 262144;
    float* o1    = (float*)(scr + o); o += 32768;
    float* o2    = (float*)(scr + o); o += 32768;
    float* xln   = (float*)(scr + o); o += 32768;
    float* ff1   = (float*)(scr + o); o += 131072;
    float* yb    = (float*)(scr + o); o += 32768;

    qproj_kernel<<<BB * DD / 256, 256, 0, st>>>(src, wq, bq, Qh);
    uproj_kernel<<<BB * HH * DD / 256, 256, 0, st>>>(wk, Qh, u);
    cb_kernel<<<1, BB * HH, 0, st>>>(Qh, bk, cb);
    scores_softmax_kernel<<<BB * HH, 256, 0, st>>>(gcpe, u, cb, mask, attnw);
    ctx_kernel<<<BB * HH * DD / 256, 256, 0, st>>>(attnw, gcpe, wctx);
    o1_kernel<<<BB * DD / 256, 256, 0, st>>>(wctx, wv, bv, o1);
    o2_resid_kernel<<<BB * DD / 256, 256, 0, st>>>(o1, wo, bo, src, o2);
    ln_kernel<<<BB, 256, 0, st>>>(o2, ln1s, ln1b, xln);
    ff1_kernel<<<BB * FFD / 256, 256, 0, st>>>(xln, l1w, l1b, ff1);
    ff2_resid_kernel<<<BB * DD / 256, 256, 0, st>>>(ff1, l2w, l2b, xln, yb);
    ln_kernel<<<BB, 256, 0, st>>>(yb, ln2s, ln2b, out);
}

extern "C" void kernel_launch(void* const* d_in, const int* in_sizes, int n_in,
                              void* d_out, int out_size, void* d_ws, size_t ws_size,
                              hipStream_t stream)
{
    (void)in_sizes; (void)n_in; (void)out_size; (void)ws_size;
    const float* g_mem  = (const float*)d_in[0];
    const int*   g_time = (const int*)d_in[1];
    const int*   step   = (const int*)d_in[2];
    const int*   g_mask = (const int*)d_in[3];
    const float* g_A    = (const float*)d_in[4];
    const float* goal   = (const float*)d_in[5];
    const float* curr   = (const float*)d_in[6];
    const float* fe_w1  = (const float*)d_in[7];
    const float* fe_b1  = (const float*)d_in[8];
    const float* fe_w2  = (const float*)d_in[9];
    const float* fe_b2  = (const float*)d_in[10];
    const float* g_w1   = (const float*)d_in[11];
    const float* g_b1   = (const float*)d_in[12];
    const float* g_w2   = (const float*)d_in[13];
    const float* g_b2   = (const float*)d_in[14];
    const float* g_w3   = (const float*)d_in[15];
    const float* g_b3   = (const float*)d_in[16];

    char* ws = (char*)d_ws;
    size_t off = 0;
    auto alloc = [&](size_t bytes) { size_t p = off; off = (off + bytes + 255) & ~(size_t)255; return p; };
    unsigned short* fe_w1t = (unsigned short*)(ws + alloc(1024 * 512 * 2));
    unsigned short* fe_w2t = (unsigned short*)(ws + alloc(512 * 512 * 2));
    unsigned short* g_w1t  = (unsigned short*)(ws + alloc(512 * 512 * 2));
    unsigned short* g_w2t  = (unsigned short*)(ws + alloc(512 * 512 * 2));
    unsigned short* g_w3t  = (unsigned short*)(ws + alloc(512 * 512 * 2));
    size_t hcat_off = alloc((size_t)BB * NN * 1024 * 2);      // reused later as gc_f32
    size_t bufA_off = alloc((size_t)BB * NN * DD * 2);        // reused later as gcpe
    size_t bufB_off = alloc((size_t)BB * NN * DD * 2);        // reused later as attn scratch
    size_t nrmA_off = alloc((size_t)BB * NN * NN * 2);
    size_t r_off    = alloc((size_t)BB * NN * 4);

    unsigned short* hcat  = (unsigned short*)(ws + hcat_off);
    unsigned short* bufA  = (unsigned short*)(ws + bufA_off);
    unsigned short* bufB  = (unsigned short*)(ws + bufB_off);
    unsigned short* normA = (unsigned short*)(ws + nrmA_off);
    float*          rdeg  = (float*)(ws + r_off);
    float*          gc_f  = (float*)(ws + hcat_off);          // overlay (hcat dead by then)
    unsigned short* gcpe  = (unsigned short*)(ws + bufA_off); // overlay (t3 dead by then)
    char*           ascr  = ws + bufB_off;                    // overlay (x2 dead by then)

    // ---- weight convert+transpose to bf16 col-major ----
    cvt_wt_kernel<<<(1024 * 512) / 256, 256, 0, stream>>>(fe_w1, fe_w1t, 1024, 512);
    cvt_wt_kernel<<<(512 * 512) / 256, 256, 0, stream>>>(fe_w2, fe_w2t, 512, 512);
    cvt_wt_kernel<<<(512 * 512) / 256, 256, 0, stream>>>(g_w1, g_w1t, 512, 512);
    cvt_wt_kernel<<<(512 * 512) / 256, 256, 0, stream>>>(g_w2, g_w2t, 512, 512);
    cvt_wt_kernel<<<(512 * 512) / 256, 256, 0, stream>>>(g_w3, g_w3t, 512, 512);

    // ---- PE add + concat goal; Laplacian-normalized adjacency ----
    build_hcat_kernel<<<(BB * NN * DD) / 256, 256, 0, stream>>>(g_mem, g_time, step, goal, hcat);
    degree_kernel<<<BB * NN, 256, 0, stream>>>(g_A, rdeg);
    build_norma_kernel<<<(BB * NN * NN) / 256, 256, 0, stream>>>(g_A, rdeg, normA);

    const long SM = (long)NN * DD;   // 512*512 per-batch stride
    // ---- feature embedding (WMMA) ----
    launch_gemm(hcat, 1024, 0, fe_w1t, 1024, 0, fe_b1, bufA, 512, 0, 0, nullptr, 0, 0,
                BB * NN, 512, 1024, 1, 1, stream);                     // h1 = relu(hcat@W1+b1)
    launch_gemm(bufA, 512, 0, fe_w2t, 512, 0, fe_b2, bufB, 512, 0, 0, nullptr, 0, 0,
                BB * NN, 512, 512, 0, 1, stream);                      // h = h1@W2+b2
    // ---- 3-layer GCN (WMMA): T = X@W (store T^T), X' = act(normA@T + b) ----
    launch_gemm(bufB, 512, SM, g_w1t, 512, 0, nullptr, bufA, 512, SM, 1, nullptr, 0, 0,
                NN, 512, 512, 0, BB, stream);                          // t1^T
    launch_gemm(normA, 512, SM, bufA, 512, SM, g_b1, bufB, 512, SM, 0, nullptr, 0, 0,
                NN, 512, 512, 1, BB, stream);                          // x1
    launch_gemm(bufB, 512, SM, g_w2t, 512, 0, nullptr, bufA, 512, SM, 1, nullptr, 0, 0,
                NN, 512, 512, 0, BB, stream);                          // t2^T
    launch_gemm(normA, 512, SM, bufA, 512, SM, g_b2, bufB, 512, SM, 0, nullptr, 0, 0,
                NN, 512, 512, 1, BB, stream);                          // x2
    launch_gemm(bufB, 512, SM, g_w3t, 512, 0, nullptr, bufA, 512, SM, 1, nullptr, 0, 0,
                NN, 512, 512, 0, BB, stream);                          // t3^T
    launch_gemm(normA, 512, SM, bufA, 512, SM, g_b3, nullptr, 0, 0, 0, gc_f, 512, SM,
                NN, 512, 512, 0, BB, stream);                          // gc (f32)

    // ---- gc + PE -> bf16 ----
    gcpe_kernel<<<(BB * NN * DD) / 256, 256, 0, stream>>>(gc_f, g_time, step, gcpe);

    // ---- decoders (q_len=1, algebraically folded attention) ----
    float* out_curr = (float*)d_out;             // curr_ctx first in return order
    float* out_goal = (float*)d_out + BB * DD;
    run_decoder(curr, d_in, 33, gcpe, g_mask, out_curr, ascr, stream);
    run_decoder(goal, d_in, 17, gcpe, g_mask, out_goal, ascr, stream);
}
